// MambaEncoder_85890755985929
// MI455X (gfx1250) — compile-verified, matrix operand order checked by reading, not run on hardware
//
#include <hip/hip_runtime.h>
#include <hip/hip_bf16.h>
#include <math.h>

// ---- model dims (match reference) ----
#define LAT   256
#define INP   120
#define NL    4
#define DST   16
#define DCONV 4
#define DIN   512
#define DTR   16
#define NG    16
#define NH    4
#define BB    32
#define TT    512
#define MTOK  (BB*TT)   // 16384 tokens

typedef __attribute__((ext_vector_type(16))) _Float16 v16h;
typedef __attribute__((ext_vector_type(8)))  float    v8f;

__device__ __forceinline__ float silu_f(float x){ return x * (1.f/(1.f+__expf(-x))); }
__device__ __forceinline__ float softplus_f(float x){ return (x > 20.f) ? x : log1pf(__expf(x)); }

// pack 4 floats -> 4 f16 -> single 64-bit LDS store (8B-aligned by construction)
__device__ __forceinline__ void pack4_store(_Float16* dst, float4 v) {
    union { _Float16 h[4]; unsigned long long u; } pk;
    pk.h[0] = (_Float16)v.x; pk.h[1] = (_Float16)v.y;
    pk.h[2] = (_Float16)v.z; pk.h[3] = (_Float16)v.w;
    *reinterpret_cast<unsigned long long*>(dst) = pk.u;
}

// CDNA5 16-bit A/B fragment K-mapping (lanes 0-15: K 0..7 / 16..23,
// lanes 16-31: K 8..15 / 24..31), pair (i, i+1) contiguous.
__device__ __forceinline__ int kmap(int i, int half) {
    return (i < 8) ? (half ? i + 8 : i) : (half ? i + 16 : i + 8);
}

// ============================================================================
// Fast WMMA GEMM (requires M%16==0, N%64==0, K%32==0):
//   C[M,N] = act(A[M,K] @ W[N,K]^T + bias[N])
// 128 threads = 4 wave32; block tile 16(M) x 64(N); wave w owns N-subtile w.
// Register double-buffered: loads for chunk k+1 issue before chunk k's WMMA,
// so s_wait_loadcnt lands after the matrix op instead of on the critical path.
// No bounds checks anywhere.
// ============================================================================
__global__ __launch_bounds__(128)
void k_gemm_fast(const float* __restrict__ A, int lda,
                 const float* __restrict__ W, int ldw,
                 const float* __restrict__ bias,
                 float* __restrict__ C, int ldc,
                 int M, int N, int K, int act)
{
    __shared__ __align__(16) _Float16 As[16][40];  // 80B rows: b64/b128-friendly
    __shared__ __align__(16) _Float16 Ws[64][40];
    const int tid  = threadIdx.x;
    const int wave = tid >> 5, lane = tid & 31;
    const int half = lane >> 4, l16 = lane & 15;
    const int m0 = blockIdx.y * 16;
    const int n0 = blockIdx.x * 64;
    const int r8 = tid >> 3, c4 = (tid & 7) << 2;

    const float* Ap = &A[(m0 + r8) * lda + c4];
    const float* Wp = &W[(n0 + r8) * ldw + c4];

    // prologue loads (chunk 0)
    float4 a_cur = *reinterpret_cast<const float4*>(Ap);
    float4 w_cur[4];
    #pragma unroll
    for (int i = 0; i < 4; ++i)
        w_cur[i] = *reinterpret_cast<const float4*>(Wp + (16 * i) * ldw);

    v8f acc = {};
    for (int kk = 0; kk < K; kk += 32) {
        // stage current chunk (f32 -> packed f16, single b64 store each)
        pack4_store(&As[r8][c4], a_cur);
        #pragma unroll
        for (int i = 0; i < 4; ++i) pack4_store(&Ws[r8 + 16 * i][c4], w_cur[i]);
        __syncthreads();
        // issue next chunk's loads now; consumed next iteration
        if (kk + 32 < K) {
            a_cur = *reinterpret_cast<const float4*>(Ap + kk + 32);
            #pragma unroll
            for (int i = 0; i < 4; ++i)
                w_cur[i] = *reinterpret_cast<const float4*>(Wp + (16 * i) * ldw + kk + 32);
            __builtin_prefetch(Ap + kk + 64, 0, 1);   // GL2 hint one chunk further
        }
        v16h af, bf;
        #pragma unroll
        for (int p = 0; p < 8; ++p) {
            int i  = 2 * p;
            int k0 = kmap(i, half);
            af[i]   = As[l16][k0];
            af[i+1] = As[l16][k0 + 1];
            bf[i]   = Ws[wave * 16 + l16][k0];
            bf[i+1] = Ws[wave * 16 + l16][k0 + 1];
        }
        acc = __builtin_amdgcn_wmma_f32_16x16x32_f16(false, af, false, bf,
                                                     (short)0, acc, false, false);
        __syncthreads();
    }

    // C/D layout: vgpr r -> (m = r + 8*half, n = lane%16); fully in-bounds.
    #pragma unroll
    for (int r = 0; r < 8; ++r) {
        int m = m0 + r + 8 * half;
        int n = n0 + wave * 16 + l16;
        float v = acc[r];
        if (bias) v += bias[n];
        if (act == 1) v = silu_f(v);
        else if (act == 2) v = softplus_f(v);
        C[m * ldc + n] = v;
    }
}

// ============================================================================
// Generic (ragged-shape) WMMA GEMM with guarded staging. Used only for
// K=120 (input proj), N=48 (x_proj), K=16 (dt_proj).
// ============================================================================
__global__ __launch_bounds__(128)
void k_gemm(const float* __restrict__ A, int lda,
            const float* __restrict__ W, int ldw,
            const float* __restrict__ bias,
            float* __restrict__ C, int ldc,
            int M, int N, int K, int act)
{
    __shared__ __align__(16) _Float16 As[16][40];
    __shared__ __align__(16) _Float16 Ws[64][40];
    const int tid  = threadIdx.x;
    const int wave = tid >> 5, lane = tid & 31;
    const int half = lane >> 4, l16 = lane & 15;
    const int m0 = blockIdx.y * 16;
    const int n0 = blockIdx.x * 64;
    const int r8 = tid >> 3, c4 = (tid & 7) << 2;

    v8f acc = {};
    for (int kk = 0; kk < K; kk += 32) {
        const bool kfull = (kk + 32 <= K);
        // ---- stage A tile 16x32 ----
        if (kfull && m0 + 16 <= M) {
            float4 av = *reinterpret_cast<const float4*>(&A[(m0 + r8) * lda + kk + c4]);
            pack4_store(&As[r8][c4], av);
        } else {
            for (int e = tid * 4; e < tid * 4 + 4; ++e) {
                int r = e >> 5, c = e & 31;
                int m = m0 + r, k = kk + c;
                float v = (m < M && k < K) ? A[m * lda + k] : 0.f;
                As[r][c] = (_Float16)v;
            }
        }
        // ---- stage W tile 64x32 ----
        if (kfull && n0 + 64 <= N) {
            #pragma unroll
            for (int i = 0; i < 4; ++i) {
                int r = r8 + 16 * i;
                float4 wv = *reinterpret_cast<const float4*>(&W[(n0 + r) * ldw + kk + c4]);
                pack4_store(&Ws[r][c4], wv);
            }
        } else {
            for (int j = 0; j < 16; ++j) {
                int e = tid * 16 + j;
                int r = e >> 5, c = e & 31;
                int n = n0 + r, k = kk + c;
                float v = (n < N && k < K) ? W[n * ldw + k] : 0.f;
                Ws[r][c] = (_Float16)v;
            }
        }
        __syncthreads();

        v16h af, bf;
        #pragma unroll
        for (int p = 0; p < 8; ++p) {
            int i  = 2 * p;
            int k0 = kmap(i, half);
            af[i]   = As[l16][k0];
            af[i+1] = As[l16][k0 + 1];
            bf[i]   = Ws[wave * 16 + l16][k0];
            bf[i+1] = Ws[wave * 16 + l16][k0 + 1];
        }
        acc = __builtin_amdgcn_wmma_f32_16x16x32_f16(false, af, false, bf,
                                                     (short)0, acc, false, false);
        __syncthreads();
    }

    #pragma unroll
    for (int r = 0; r < 8; ++r) {
        int m = m0 + r + 8 * half;
        int n = n0 + wave * 16 + l16;
        if (m < M && n < N) {
            float v = acc[r];
            if (bias) v += bias[n];
            if (act == 1) v = silu_f(v);
            else if (act == 2) v = softplus_f(v);
            C[m * ldc + n] = v;
        }
    }
}

// ============================================================================
// Local-module conv (k=3, same-pad) as WMMA: out[t,co] =
//   silu(sum_dt sum_ci w[co,ci,dt]*gn[t+dt-1,ci] + b[co]) + res, masked.
// Loops 3 taps x 8 K-chunks into one accumulator. 16-token tiles never
// straddle a batch (T=512 % 16 == 0).
// ============================================================================
__global__ __launch_bounds__(128)
void k_conv3(const float* __restrict__ gn, const float* __restrict__ res,
             const float* __restrict__ w, const float* __restrict__ bias,
             const int* __restrict__ len, float* __restrict__ out)
{
    __shared__ __align__(16) _Float16 As[16][40];
    __shared__ __align__(16) _Float16 Ws[64][40];
    const int tid  = threadIdx.x;
    const int wave = tid >> 5, lane = tid & 31;
    const int half = lane >> 4, l16 = lane & 15;
    const int m0 = blockIdx.y * 16;   // token tile
    const int n0 = blockIdx.x * 64;   // out-channel tile
    const int r8 = tid >> 3, c4 = (tid & 7) << 2;

    v8f acc = {};
    for (int dt = 0; dt < 3; ++dt) {
        for (int kk = 0; kk < LAT; kk += 32) {
            // stage shifted activations: row r -> token m0+r, time t+dt-1
            {
                int tok = m0 + r8;
                int tt  = (tok & (TT - 1)) + dt - 1;
                float4 av = make_float4(0.f, 0.f, 0.f, 0.f);
                if (tt >= 0 && tt < TT)
                    av = *reinterpret_cast<const float4*>(&gn[(tok + dt - 1) * LAT + kk + c4]);
                pack4_store(&As[r8][c4], av);
            }
            // stage weight slice w[n][k][dt] (stride-3 gather, branchless)
            for (int j = 0; j < 16; ++j) {
                int e = tid * 16 + j;
                int r = e >> 5, c = e & 31;
                Ws[r][c] = (_Float16)w[(n0 + r) * (LAT * 3) + (kk + c) * 3 + dt];
            }
            __syncthreads();

            v16h af, bf;
            #pragma unroll
            for (int p = 0; p < 8; ++p) {
                int i  = 2 * p;
                int k0 = kmap(i, half);
                af[i]   = As[l16][k0];
                af[i+1] = As[l16][k0 + 1];
                bf[i]   = Ws[wave * 16 + l16][k0];
                bf[i+1] = Ws[wave * 16 + l16][k0 + 1];
            }
            acc = __builtin_amdgcn_wmma_f32_16x16x32_f16(false, af, false, bf,
                                                         (short)0, acc, false, false);
            __syncthreads();
        }
    }

    #pragma unroll
    for (int r = 0; r < 8; ++r) {
        int m = m0 + r + 8 * half;           // token index
        int n = n0 + wave * 16 + l16;        // out channel
        int t = m & (TT - 1), bb = m >> 9;   // TT == 512
        float v = silu_f(acc[r] + bias[n]) + res[m * LAT + n];
        out[m * LAT + n] = (t < len[bb]) ? v : 0.f;
    }
}

// ---- LayerNorm over C=256 (one block per token row), optional SiLU + pos ----
__global__ __launch_bounds__(256)
void k_layernorm(const float* __restrict__ in, float* __restrict__ out,
                 const float* __restrict__ g, const float* __restrict__ b,
                 const float* __restrict__ pos, int do_silu)
{
    __shared__ float red[256];
    int row = blockIdx.x, c = threadIdx.x;
    float v = in[row * LAT + c];
    red[c] = v; __syncthreads();
    for (int s = 128; s > 0; s >>= 1) { if (c < s) red[c] += red[c + s]; __syncthreads(); }
    float mean = red[0] * (1.f / LAT);
    __syncthreads();
    float d = v - mean;
    red[c] = d * d; __syncthreads();
    for (int s = 128; s > 0; s >>= 1) { if (c < s) red[c] += red[c + s]; __syncthreads(); }
    float var = red[0] * (1.f / LAT);
    float o = d * rsqrtf(var + 1e-5f) * g[c] + b[c];
    if (do_silu) o = silu_f(o);
    if (pos) o += pos[(row % TT) * LAT + c];
    out[row * LAT + c] = o;
}

// ---- GroupNorm: one block per (batch, group); stats over 16 ch x 512 t ----
__global__ __launch_bounds__(256)
void k_groupnorm(const float* __restrict__ h, float* __restrict__ out,
                 const float* __restrict__ g, const float* __restrict__ b)
{
    __shared__ float rs[256], rs2[256];
    int bg = blockIdx.x; int bb = bg / NG, grp = bg % NG;
    int tid = threadIdx.x;
    float s = 0.f, s2 = 0.f;
    for (int e = tid; e < 16 * TT; e += 256) {
        int ch = grp * 16 + (e & 15), t = e >> 4;
        float v = h[(bb * TT + t) * LAT + ch];
        s += v; s2 += v * v;
    }
    rs[tid] = s; rs2[tid] = s2; __syncthreads();
    for (int k = 128; k > 0; k >>= 1) {
        if (tid < k) { rs[tid] += rs[tid + k]; rs2[tid] += rs2[tid + k]; }
        __syncthreads();
    }
    float mean = rs[0] / (16.f * TT);
    float var  = rs2[0] / (16.f * TT) - mean * mean;
    float inv  = rsqrtf(var + 1e-5f);
    for (int e = tid; e < 16 * TT; e += 256) {
        int ch = grp * 16 + (e & 15), t = e >> 4;
        int idx = (bb * TT + t) * LAT + ch;
        out[idx] = (h[idx] - mean) * inv * g[ch] + b[ch];
    }
}

// ---- time-flip copy: dst[b, T-1-t, coloff+c] = src[b, t, c] ----
__global__ void k_flip(const float* __restrict__ src, float* __restrict__ dst,
                       int C, int ldd, int coloff)
{
    int idx = blockIdx.x * 256 + threadIdx.x;     // over MTOK*C
    int c = idx % C; int bt = idx / C;
    int t = bt % TT, bb = bt / TT;
    dst[(bb * TT + (TT - 1 - t)) * ldd + coloff + c] = src[idx];
}

// ---- causal depthwise conv (d_conv=4) over xi (cols 0..511 of xz) + SiLU ----
__global__ void k_dwconv(const float* __restrict__ xz, const float* __restrict__ w,
                         const float* __restrict__ bias, float* __restrict__ xc)
{
    int idx = blockIdx.x * 256 + threadIdx.x;     // over MTOK*DIN
    int d = idx % DIN; int bt = idx / DIN;
    int t = bt % TT, bb = bt / TT;
    float acc = bias[d];
    #pragma unroll
    for (int j = 0; j < DCONV; ++j) {
        int tt = t + j - (DCONV - 1);
        if (tt >= 0) acc += w[d * DCONV + j] * xz[(bb * TT + tt) * (2 * DIN) + d];
    }
    xc[idx] = silu_f(acc);
}

// ---- selective scan: 1 block per batch, 512 threads (one per channel d).
// 16-wide SSM state in registers; per-step B/C staged through LDS. ----
__global__ __launch_bounds__(512)
void k_scan(const float* __restrict__ xc, const float* __restrict__ delta,
            const float* __restrict__ dbc, const float* __restrict__ A_log,
            const float* __restrict__ D, float* __restrict__ out)
{
    __shared__ float sB[DST], sC[DST];
    int bb = blockIdx.x, d = threadIdx.x;
    float Av[DST], st[DST];
    #pragma unroll
    for (int n = 0; n < DST; ++n) { Av[n] = -__expf(A_log[d * DST + n]); st[n] = 0.f; }
    float Dd = D[d];
    for (int t = 0; t < TT; ++t) {
        int row = bb * TT + t;
        if (d < DST)            sB[d]       = dbc[row * 48 + DTR + d];
        else if (d < 2 * DST)   sC[d - DST] = dbc[row * 48 + DTR + DST + (d - DST)];
        __syncthreads();
        float u  = xc[row * DIN + d];
        float dl = delta[row * DIN + d];
        float du = dl * u, y = 0.f;
        #pragma unroll
        for (int n = 0; n < DST; ++n) {
            float dA = __expf(dl * Av[n]);
            st[n] = dA * st[n] + du * sB[n];
            y += st[n] * sC[n];
        }
        out[row * DIN + d] = y + u * Dd;
        __syncthreads();
    }
}

// ---- y = scan_out * silu(z)  (z = cols 512..1023 of xz) ----
__global__ void k_gate(const float* __restrict__ sc, const float* __restrict__ xz,
                       float* __restrict__ y)
{
    int idx = blockIdx.x * 256 + threadIdx.x;     // over MTOK*DIN
    int d = idx % DIN; int bt = idx / DIN;
    float z = xz[bt * (2 * DIN) + DIN + d];
    y[idx] = sc[idx] * silu_f(z);
}

// ---- h = (a + h) * mask ----
__global__ void k_addres_mask(const float* __restrict__ a, float* __restrict__ h,
                              const int* __restrict__ len)
{
    int idx = blockIdx.x * 256 + threadIdx.x;     // over MTOK*LAT
    int bt = idx / LAT; int t = bt % TT, bb = bt / TT;
    float v = a[idx] + h[idx];
    h[idx] = (t < len[bb]) ? v : 0.f;
}

// ---- q projection (query is batch-independent): qp = Wq @ pool_q + bq ----
__global__ __launch_bounds__(256)
void k_qproj(const float* __restrict__ q, const float* __restrict__ Wq,
             const float* __restrict__ bq, float* __restrict__ qp)
{
    int c = threadIdx.x;
    float acc = bq[c];
    for (int k = 0; k < LAT; ++k) acc += Wq[c * LAT + k] * q[k];
    qp[c] = acc;
}

// ---- masked-softmax attention pooling (1 query, 4 heads); block per batch ----
__global__ __launch_bounds__(256)
void k_pool(const float* __restrict__ qp, const float* __restrict__ kp,
            const float* __restrict__ vp, const int* __restrict__ len,
            float* __restrict__ out)
{
    __shared__ float sc[NH * TT];
    __shared__ float red[256];
    int bb = blockIdx.x, tid = threadIdx.x;
    int hd = tid >> 6, j = tid & 63;
    int L = len[bb];
    const float scale = 0.125f;                      // 1/sqrt(64)
    for (int t = j; t < TT; t += 64) {
        float s = 0.f;
        const float* kr = &kp[(bb * TT + t) * LAT + hd * 64];
        for (int dd = 0; dd < 64; ++dd) s += qp[hd * 64 + dd] * kr[dd];
        sc[hd * TT + t] = (t < L) ? s * scale : -1e30f;
    }
    __syncthreads();
    float lm = -1e30f;
    for (int t = j; t < TT; t += 64) lm = fmaxf(lm, sc[hd * TT + t]);
    red[tid] = lm; __syncthreads();
    for (int k = 32; k > 0; k >>= 1) { if (j < k) red[tid] = fmaxf(red[tid], red[tid + k]); __syncthreads(); }
    float mx = red[hd * 64];
    __syncthreads();
    float ls = 0.f;
    for (int t = j; t < TT; t += 64) { float e = __expf(sc[hd * TT + t] - mx); sc[hd * TT + t] = e; ls += e; }
    red[tid] = ls; __syncthreads();
    for (int k = 32; k > 0; k >>= 1) { if (j < k) red[tid] += red[tid + k]; __syncthreads(); }
    float inv = 1.f / red[hd * 64];
    __syncthreads();
    float o = 0.f;
    int d = hd * 64 + j;
    for (int t = 0; t < TT; ++t) o += sc[hd * TT + t] * vp[(bb * TT + t) * LAT + d];
    out[bb * LAT + d] = o * inv;
}

// ============================================================================
// Host side
// ============================================================================
struct MP { const float *in_w,*conv_w,*conv_b,*xproj_w,*dt_w,*dt_b,*A_log,*D,*out_w; };
struct LayerP {
    const float *gn_g,*gn_b,*lconv_w,*lconv_b,*ln_g,*ln_b;
    MP fwd, bwd;
    const float *comb_w,*comb_b;
};

static inline float* carve(char*& p, size_t nfloats) {
    float* r = (float*)p; p += nfloats * sizeof(float); return r;
}

static void run_gemm(const float* A, int lda, const float* W, int ldw,
                     const float* bias, float* C, int ldc,
                     int M, int N, int K, int act, hipStream_t s)
{
    dim3 grid((N + 63) / 64, (M + 15) / 16);
    if ((M % 16) == 0 && (N % 64) == 0 && (K % 32) == 0)
        k_gemm_fast<<<grid, 128, 0, s>>>(A, lda, W, ldw, bias, C, ldc, M, N, K, act);
    else
        k_gemm<<<grid, 128, 0, s>>>(A, lda, W, ldw, bias, C, ldc, M, N, K, act);
}

extern "C" void kernel_launch(void* const* d_in, const int* in_sizes, int n_in,
                              void* d_out, int out_size, void* d_ws, size_t ws_size,
                              hipStream_t stream)
{
    (void)in_sizes; (void)n_in; (void)out_size; (void)ws_size;

    // ---- walk inputs in setup_inputs / make_params insertion order ----
    int ip = 0;
    const float* x       = (const float*)d_in[ip++];
    const int*   lengths = (const int*)  d_in[ip++];
    auto nxt = [&]() { return (const float*)d_in[ip++]; };
    const float *in_w = nxt(), *in_b = nxt(), *in_ln_g = nxt(), *in_ln_b = nxt(), *pos = nxt();
    LayerP layers[NL];
    for (int l = 0; l < NL; ++l) {
        LayerP& L = layers[l];
        L.gn_g = nxt(); L.gn_b = nxt(); L.lconv_w = nxt(); L.lconv_b = nxt();
        L.ln_g = nxt(); L.ln_b = nxt();
        auto rdmp = [&](MP& m) {
            m.in_w = nxt(); m.conv_w = nxt(); m.conv_b = nxt(); m.xproj_w = nxt();
            m.dt_w = nxt(); m.dt_b = nxt(); m.A_log = nxt(); m.D = nxt(); m.out_w = nxt();
        };
        rdmp(L.fwd); rdmp(L.bwd);
        L.comb_w = nxt(); L.comb_b = nxt();
    }
    const float *fin_g = nxt(), *fin_b = nxt(), *pool_q = nxt();
    const float *attn_in_w = nxt(), *attn_in_b = nxt();
    const float *attn_out_w = nxt(), *attn_out_b = nxt();
    const float *mu_w = nxt(), *mu_b = nxt(), *lv_w = nxt(), *lv_b = nxt();

    // ---- scratch carving ----
    char* wsp = (char*)d_ws;
    float* tmp    = carve(wsp, (size_t)MTOK * LAT);
    float* h      = carve(wsp, (size_t)MTOK * LAT);
    float* gbuf   = carve(wsp, (size_t)MTOK * LAT);
    float* lnbuf  = carve(wsp, (size_t)MTOK * LAT);
    float* hflip  = carve(wsp, (size_t)MTOK * LAT);
    float* xz     = carve(wsp, (size_t)MTOK * 2 * DIN);
    float* xc     = carve(wsp, (size_t)MTOK * DIN);
    float* dbc    = carve(wsp, (size_t)MTOK * 48);
    float* delta  = carve(wsp, (size_t)MTOK * DIN);
    float* scanb  = carve(wsp, (size_t)MTOK * DIN);
    float* ybuf   = carve(wsp, (size_t)MTOK * DIN);
    float* cat    = carve(wsp, (size_t)MTOK * 2 * LAT);
    float* dtmp   = carve(wsp, (size_t)MTOK * LAT);
    float* mbout  = carve(wsp, (size_t)MTOK * LAT);
    float* finb   = carve(wsp, (size_t)MTOK * LAT);
    float* kp     = carve(wsp, (size_t)MTOK * LAT);
    float* vp     = carve(wsp, (size_t)MTOK * LAT);
    float* qp     = carve(wsp, LAT);
    float* attn_o = carve(wsp, (size_t)BB * LAT);
    float* pooled = carve(wsp, (size_t)BB * LAT);

    const int EW256 = MTOK * LAT / 256;
    const int EW512 = MTOK * DIN / 256;

    // ---- input stage: h = silu(LN(x @ in_w^T + in_b)) + pos ----
    run_gemm(x, INP, in_w, INP, in_b, tmp, LAT, MTOK, LAT, INP, 0, stream);
    k_layernorm<<<MTOK, 256, 0, stream>>>(tmp, h, in_ln_g, in_ln_b, pos, 1);

    // ---- one Mamba direction ----
    auto mamba_dir = [&](const float* hin, const MP& m, float* dirout, int ldc_out) {
        run_gemm(hin, LAT, m.in_w, LAT, nullptr, xz, 2 * DIN, MTOK, 2 * DIN, LAT, 0, stream);
        k_dwconv<<<EW512, 256, 0, stream>>>(xz, m.conv_w, m.conv_b, xc);
        run_gemm(xc, DIN, m.xproj_w, DIN, nullptr, dbc, 48, MTOK, 48, DIN, 0, stream);
        run_gemm(dbc, 48, m.dt_w, DTR, m.dt_b, delta, DIN, MTOK, DIN, DTR, 2, stream); // softplus
        k_scan<<<BB, DIN, 0, stream>>>(xc, delta, dbc, m.A_log, m.D, scanb);
        k_gate<<<EW512, 256, 0, stream>>>(scanb, xz, ybuf);
        run_gemm(ybuf, DIN, m.out_w, DIN, nullptr, dirout, ldc_out, MTOK, LAT, DIN, 0, stream);
    };

    // ---- layers ----
    for (int l = 0; l < NL; ++l) {
        const LayerP& L = layers[l];
        // local module (WMMA conv)
        k_groupnorm<<<BB * NG, 256, 0, stream>>>(h, gbuf, L.gn_g, L.gn_b);
        k_conv3<<<dim3(LAT / 64, MTOK / 16), 128, 0, stream>>>(
            gbuf, h, L.lconv_w, L.lconv_b, lengths, h);
        // mamba block
        k_layernorm<<<MTOK, 256, 0, stream>>>(h, lnbuf, L.ln_g, L.ln_b, nullptr, 0);
        mamba_dir(lnbuf, L.fwd, cat, 2 * LAT);                    // fwd -> cat[:, 0:256]
        k_flip<<<EW256, 256, 0, stream>>>(lnbuf, hflip, LAT, LAT, 0);
        mamba_dir(hflip, L.bwd, dtmp, LAT);                       // bwd on flipped h
        k_flip<<<EW256, 256, 0, stream>>>(dtmp, cat, LAT, 2 * LAT, LAT); // unflip -> cat[:,256:]
        run_gemm(cat, 2 * LAT, L.comb_w, 2 * LAT, L.comb_b, mbout, LAT,
                 MTOK, LAT, 2 * LAT, 0, stream);
        k_addres_mask<<<EW256, 256, 0, stream>>>(mbout, h, lengths);
    }

    // ---- final LN + attention pooling + heads ----
    k_layernorm<<<MTOK, 256, 0, stream>>>(h, finb, fin_g, fin_b, nullptr, 0);
    run_gemm(finb, LAT, attn_in_w + LAT * LAT,     LAT, attn_in_b + LAT,     kp, LAT,
             MTOK, LAT, LAT, 0, stream);                          // K projection
    run_gemm(finb, LAT, attn_in_w + 2 * LAT * LAT, LAT, attn_in_b + 2 * LAT, vp, LAT,
             MTOK, LAT, LAT, 0, stream);                          // V projection
    k_qproj<<<1, 256, 0, stream>>>(pool_q, attn_in_w, attn_in_b, qp);
    k_pool<<<BB, 256, 0, stream>>>(qp, kp, vp, lengths, attn_o);
    run_gemm(attn_o, LAT, attn_out_w, LAT, attn_out_b, pooled, LAT, BB, LAT, LAT, 0, stream);

    float* out = (float*)d_out;
    run_gemm(pooled, LAT, mu_w, LAT, mu_b, out,            LAT, BB, LAT, LAT, 0, stream); // mu
    run_gemm(pooled, LAT, lv_w, LAT, lv_b, out + BB * LAT, LAT, BB, LAT, LAT, 0, stream); // logvar
}
